// RNNBlock_34660386079343
// MI455X (gfx1250) — matrix operand order repaired and checked by reading, not compiled, not run on hardware
//
#include <hip/hip_runtime.h>
#include <hip/hip_bf16.h>

// ---------------- Mamba2 dims (match reference) ----------------
#define DMODEL   512
#define DSTATE   128
#define DINNER   1024
#define HEADDIM  64
#define NHEADS   16
#define CONVDIM  1280
#define DINPROJ  2320
#define DINPROJ_PAD 2432             // 19 * 128, zero-padded bf16 weight rows
#define TSEQ     8
#define BFLAT    256                 // B*H*W
#define NROWS    (BFLAT*TSEQ)        // 2048 flattened (batch,t) rows

typedef __attribute__((ext_vector_type(16))) __bf16 v16bf;
typedef __attribute__((ext_vector_type(8)))  float  v8f;

__device__ __forceinline__ v8f zero8() {
  v8f z = {0.f,0.f,0.f,0.f,0.f,0.f,0.f,0.f};
  return z;
}
__device__ __forceinline__ float siluf(float v)     { return v * __frcp_rn(1.f + __expf(-v)); }
__device__ __forceinline__ float softplusf(float v) { return (v > 20.f) ? v : __logf(1.f + __expf(v)); }

// x is (1, T=8, H=16, W=16, 512); flattened row r = bf*8 + t, bf = h*16 + w
__device__ __forceinline__ size_t x_index(int r, int d) {
  int t = r & 7, w = (r >> 3) & 15, h = r >> 7;
  return ((size_t)(((t * 16 + h) * 16 + w)) << 9) + d;
}

// ---------------- CDNA5 async global->LDS copy (ASYNCcnt-tracked) ----------------
__device__ __forceinline__ void async_ld_b128(void* lds_ptr, const void* gptr) {
  unsigned lds_off = (unsigned)(unsigned long long)lds_ptr;  // low 32 bits = LDS byte offset
  asm volatile("global_load_async_to_lds_b128 %0, %1, off"
               :: "v"(lds_off), "v"(gptr)
               : "memory");
}
__device__ __forceinline__ void wait_async0() {
#if __has_builtin(__builtin_amdgcn_s_wait_asynccnt)
  __builtin_amdgcn_s_wait_asynccnt(0);
#else
  asm volatile("s_wait_asynccnt 0x0" ::: "memory");
#endif
}

// ---------------- WMMA fragment helpers (16x16x32 bf16) ----------------
__device__ __forceinline__ v16bf load_a_frag(const __bf16* base, int row0, int lane,
                                             int ldk, int k0) {
  const int half = lane >> 4;
  const int mr   = lane & 15;
  const __bf16* p = base + (size_t)(row0 + mr) * ldk + k0;
  v16bf a;
#pragma unroll
  for (int e = 0; e < 8; ++e) a[e] = p[half * 8 + e];
#pragma unroll
  for (int e = 0; e < 8; ++e) a[8 + e] = p[16 + half * 8 + e];
  return a;
}
__device__ __forceinline__ v16bf load_b_frag(const __bf16* base, int col0, int lane,
                                             int ldk, int k0) {
  const int half = lane >> 4;
  const int nr   = lane & 15;
  const __bf16* p = base + (size_t)(col0 + nr) * ldk + k0;
  v16bf b;
#pragma unroll
  for (int e = 0; e < 8; ++e) b[e] = p[half * 8 + e];
#pragma unroll
  for (int e = 0; e < 8; ++e) b[8 + e] = p[16 + half * 8 + e];
  return b;
}
__device__ __forceinline__ v8f wmma_bf16(v16bf a, v16bf b, v8f c) {
  return __builtin_amdgcn_wmma_f32_16x16x32_bf16(false, a, false, b, (short)0, c,
                                                 false, false);
}

union pk8 { __bf16 h[8]; uint4 u; };
union pk4 { __bf16 h[4]; uint2 u; };

// ================= Kernel 0: one-shot f32 -> bf16 weight conversion =================
__global__ __launch_bounds__(256) void k0_cvt(const float* __restrict__ src,
                                              __bf16* __restrict__ dst,
                                              int nsrc, int ntot) {
  int e = (blockIdx.x * 256 + threadIdx.x) * 8;
  if (e >= ntot) return;
  pk8 pk;
  if (e + 8 <= nsrc) {
    float4 f0 = ((const float4*)(src + e))[0];
    float4 f1 = ((const float4*)(src + e))[1];
    pk.h[0] = (__bf16)f0.x; pk.h[1] = (__bf16)f0.y;
    pk.h[2] = (__bf16)f0.z; pk.h[3] = (__bf16)f0.w;
    pk.h[4] = (__bf16)f1.x; pk.h[5] = (__bf16)f1.y;
    pk.h[6] = (__bf16)f1.z; pk.h[7] = (__bf16)f1.w;
  } else {
#pragma unroll
    for (int j = 0; j < 8; ++j) pk.h[j] = (__bf16)((e + j < nsrc) ? src[e + j] : 0.f);
  }
  *(uint4*)(dst + e) = pk.u;
}

// ================= Kernel 1: in_proj GEMM (double-buffered async pipeline) ==========
// zx[r][n] = sum_d X[r][d] * W[n][d], M=2048, N=2320(->2432), K=512. BM=64, BN=128.
__global__ __launch_bounds__(256) void k1_inproj(const float* __restrict__ x,
                                                 const __bf16* __restrict__ Wi,
                                                 float* __restrict__ zx) {
  __shared__ __bf16 aT[2][64 * 32];
  __shared__ __bf16 bT[2][128 * 32];
  const int tid  = threadIdx.x;
  const int lane = tid & 31;
  const int wid  = tid >> 5;
  const int wm   = wid >> 2;            // 0..1
  const int wn   = wid & 3;             // 0..3
  const int rBase = blockIdx.x * 64;
  const int nBase = blockIdx.y * 128;
  const int half = lane >> 4, nr = lane & 15;

  const int arow = tid >> 2, aseg = (tid & 3) * 8;   // A: 8 contiguous k per thread
  const int brow = tid >> 1, bseg = (tid & 1) * 16;  // B: 32 bytes per thread

  v8f acc[2][2] = {{zero8(), zero8()}, {zero8(), zero8()}};

  auto stage = [&](int k0, int buf) {
    // B tile: pure bf16 copy -> async DMA into LDS (rows zero-padded, no guard)
    const __bf16* bsrc = Wi + (size_t)(nBase + brow) * DMODEL + k0 + bseg;
    async_ld_b128(&bT[buf][brow * 32 + bseg], bsrc);
    async_ld_b128(&bT[buf][brow * 32 + bseg + 8], bsrc + 8);
    if (k0 + 32 < DMODEL)
      __builtin_prefetch(bsrc + 32, 0, 1);
    // A tile: gather from x + f32->bf16, vectorized (2x float4 -> 1x b128 store)
    const float* src = x + x_index(rBase + arow, k0 + aseg);
    float4 f0 = ((const float4*)src)[0];
    float4 f1 = ((const float4*)src)[1];
    pk8 pk;
    pk.h[0] = (__bf16)f0.x; pk.h[1] = (__bf16)f0.y;
    pk.h[2] = (__bf16)f0.z; pk.h[3] = (__bf16)f0.w;
    pk.h[4] = (__bf16)f1.x; pk.h[5] = (__bf16)f1.y;
    pk.h[6] = (__bf16)f1.z; pk.h[7] = (__bf16)f1.w;
    *(uint4*)&aT[buf][arow * 32 + aseg] = pk.u;
  };

  stage(0, 0);
  const int NK = DMODEL / 32;           // 16
  for (int kt = 0; kt < NK; ++kt) {
    wait_async0();                      // own copies for tile kt done
    __syncthreads();                    // -> all waves' tile-kt data resident
    if (kt + 1 < NK) stage((kt + 1) * 32, (kt + 1) & 1);   // overlap next tile
    const __bf16* aB = aT[kt & 1];
    const __bf16* bB = bT[kt & 1];
    v16bf af0 = load_a_frag(aB, wm * 32,      lane, 32, 0);
    v16bf af1 = load_a_frag(aB, wm * 32 + 16, lane, 32, 0);
    v16bf bf0 = load_b_frag(bB, wn * 32,      lane, 32, 0);
    v16bf bf1 = load_b_frag(bB, wn * 32 + 16, lane, 32, 0);
    acc[0][0] = wmma_bf16(af0, bf0, acc[0][0]);
    acc[0][1] = wmma_bf16(af0, bf1, acc[0][1]);
    acc[1][0] = wmma_bf16(af1, bf0, acc[1][0]);
    acc[1][1] = wmma_bf16(af1, bf1, acc[1][1]);
  }
#pragma unroll
  for (int mi = 0; mi < 2; ++mi)
#pragma unroll
    for (int ni = 0; ni < 2; ++ni)
#pragma unroll
      for (int i = 0; i < 8; ++i) {
        int r  = rBase + wm * 32 + mi * 16 + half * 8 + i;
        int gn = nBase + wn * 32 + ni * 16 + nr;
        if (gn < DINPROJ) zx[(size_t)r * DINPROJ + gn] = acc[mi][ni][i];
      }
}

// ================= Kernel 2: depthwise causal conv + SiLU =================
__global__ __launch_bounds__(256) void k2_conv(const float* __restrict__ zx,
                                               const float* __restrict__ cw,
                                               const float* __restrict__ cb,
                                               float* __restrict__ xBC) {
  int g = blockIdx.x * 256 + threadIdx.x;
  int c = g % CONVDIM, bf = g / CONVDIM;
  float w0 = cw[c * 4 + 0], w1 = cw[c * 4 + 1], w2 = cw[c * 4 + 2], w3 = cw[c * 4 + 3];
  float b = cb[c];
  float s0 = 0.f, s1 = 0.f, s2 = 0.f;
#pragma unroll
  for (int t = 0; t < TSEQ; ++t) {
    float xin = zx[(size_t)(bf * TSEQ + t) * DINPROJ + DINNER + c];
    float v = fmaf(s0, w0, fmaf(s1, w1, fmaf(s2, w2, fmaf(xin, w3, b))));
    xBC[(size_t)(bf * TSEQ + t) * CONVDIM + c] = siluf(v);
    s0 = s1; s1 = s2; s2 = xin;
  }
}

// ================= Kernel 2b: dt = softplus(.), dA = exp(dt*A) =================
__global__ __launch_bounds__(256) void k2b_dt(const float* __restrict__ zx,
                                              const float* __restrict__ dt_bias,
                                              const float* __restrict__ A_log,
                                              float* __restrict__ dtb,
                                              float* __restrict__ dAb) {
  int g = blockIdx.x * 256 + threadIdx.x;
  int r = g >> 4, hh = g & 15;
  float dt = softplusf(zx[(size_t)r * DINPROJ + (DINNER + CONVDIM) + hh] + dt_bias[hh]);
  float dA = __expf(-__expf(A_log[hh]) * dt);
  dtb[g] = dt;
  dAb[g] = dA;
}

// ================= Kernel 3: SSD core, one (bf, head) pair per wave =================
__global__ __launch_bounds__(128) void k3_ssd(const float* __restrict__ xBC,
                                              const float* __restrict__ dtb,
                                              const float* __restrict__ dAb,
                                              const float* __restrict__ Dvec,
                                              float* __restrict__ ybuf) {
  __shared__ __bf16 cS[4][16 * 128];
  __shared__ __bf16 bS[4][16 * 128];
  __shared__ __bf16 xS[4][64 * 32];
  __shared__ float  gS[4][16 * 16];
  const int lane = threadIdx.x & 31;
  const int wv   = threadIdx.x >> 5;
  const int pair = blockIdx.x * 4 + wv;
  const int bf   = pair >> 4;
  const int h    = pair & 15;
  const int half = lane >> 4, nr = lane & 15;
  __bf16* cL = cS[wv];
  __bf16* bL = bS[wv];
  __bf16* xL = xS[wv];
  float*  gL = gS[wv];

  // Stage C / B tiles: one 128-wide row per step, lane covers 4 contiguous k.
#pragma unroll
  for (int t = 0; t < 16; ++t) {
    const int k4 = lane * 4;
    float4 cv = {0.f,0.f,0.f,0.f}, bv = {0.f,0.f,0.f,0.f};
    if (t < TSEQ) {
      const float* row = xBC + (size_t)(bf * TSEQ + t) * CONVDIM;
      cv = *(const float4*)(row + DINNER + DSTATE + k4);   // Cm
      bv = *(const float4*)(row + DINNER + k4);            // Bm
    }
    pk4 pc, pb;
    pc.h[0] = (__bf16)cv.x; pc.h[1] = (__bf16)cv.y;
    pc.h[2] = (__bf16)cv.z; pc.h[3] = (__bf16)cv.w;
    pb.h[0] = (__bf16)bv.x; pb.h[1] = (__bf16)bv.y;
    pb.h[2] = (__bf16)bv.z; pb.h[3] = (__bf16)bv.w;
    *(uint2*)&cL[t * 128 + k4] = pc.u;
    *(uint2*)&bL[t * 128 + k4] = pb.u;
  }
  // Stage X head slice (64 x 32 K, s>=8 zero): zero-fill then write valid entries.
  {
    uint4 z4 = {0u,0u,0u,0u};
#pragma unroll
    for (int i = 0; i < 8; ++i) *(uint4*)&xL[i * 256 + lane * 8] = z4;
#pragma unroll
    for (int j = 0; j < 16; ++j) {
      int idx = j * 32 + lane;            // 512 valid elems: p = idx/8, s = idx&7
      int p = idx >> 3, s = idx & 7;
      float xv = xBC[(size_t)(bf * TSEQ + s) * CONVDIM + h * HEADDIM + p];
      xL[p * 32 + s] = (__bf16)xv;
    }
  }
  __syncthreads();

  // G = C @ B^T over d_state = 128
  v8f g = zero8();
#pragma unroll
  for (int kt = 0; kt < 4; ++kt) {
    v16bf a = load_a_frag(cL, 0, lane, 128, kt * 32);
    v16bf b = load_b_frag(bL, 0, lane, 128, kt * 32);
    g = wmma_bf16(a, b, g);
  }
#pragma unroll
  for (int i = 0; i < 8; ++i) gL[(half * 8 + i) * 16 + nr] = g[i];
  __syncthreads();

  float dtv[TSEQ], Pv[TSEQ];
  float cum = 1.f;
#pragma unroll
  for (int s = 0; s < TSEQ; ++s) {
    dtv[s] = dtb[(bf * TSEQ + s) * NHEADS + h];
    cum   *= dAb[(bf * TSEQ + s) * NHEADS + h];
    Pv[s]  = cum;
  }

  // Masked scores as an A fragment: M[t,s] = G[t,s]*dt_s*P[t]/P[s] for s<=t<8.
  v16bf ma;
  const int t = nr;
#pragma unroll
  for (int e = 0; e < 16; ++e) {
    int k = (e < 8) ? (half * 8 + e) : (16 + half * 8 + (e - 8));
    float v = 0.f;
    if (t < TSEQ && k < TSEQ && k <= t)
      v = gL[t * 16 + k] * dtv[k] * (Pv[t] * __frcp_rn(Pv[k]));
    ma[e] = (__bf16)v;
  }

  const float Dh = Dvec[h];
#pragma unroll
  for (int nt = 0; nt < 4; ++nt) {
    v16bf xb = load_b_frag(xL, nt * 16, lane, 32, 0);
    v8f y = wmma_bf16(ma, xb, zero8());
    if (half == 0) {
#pragma unroll
      for (int i = 0; i < 8; ++i) {
        int p = nt * 16 + nr;
        float xv = xBC[(size_t)(bf * TSEQ + i) * CONVDIM + h * HEADDIM + p];
        ybuf[(size_t)(bf * TSEQ + i) * DINNER + h * HEADDIM + p] = y[i] + Dh * xv;
      }
    }
  }
}

// ================= Kernel 4: gate (y * silu(z)) + RMSNorm -> bf16 =================
__global__ __launch_bounds__(256) void k4_gatenorm(const float* __restrict__ zx,
                                                   const float* __restrict__ ybuf,
                                                   const float* __restrict__ norm_w,
                                                   __bf16* __restrict__ ynorm) {
  __shared__ float red[256];
  const int r = blockIdx.x, tid = threadIdx.x;
  float gv[4], ss = 0.f;
#pragma unroll
  for (int j = 0; j < 4; ++j) {
    int c = tid + j * 256;
    float y = ybuf[(size_t)r * DINNER + c];
    float z = zx[(size_t)r * DINPROJ + c];
    float v = y * siluf(z);
    gv[j] = v;
    ss += v * v;
  }
  red[tid] = ss;
  __syncthreads();
  for (int s = 128; s > 0; s >>= 1) {
    if (tid < s) red[tid] += red[tid + s];
    __syncthreads();
  }
  float scale = rsqrtf(red[0] * (1.f / (float)DINNER) + 1e-5f);
#pragma unroll
  for (int j = 0; j < 4; ++j) {
    int c = tid + j * 256;
    ynorm[(size_t)r * DINNER + c] = (__bf16)(gv[j] * scale * norm_w[c]);
  }
}

// ================= Kernel 5: out_proj GEMM (double-buffered async) + scatter ========
__global__ __launch_bounds__(256) void k5_outproj(const __bf16* __restrict__ ynorm,
                                                  const __bf16* __restrict__ Wo,
                                                  float* __restrict__ out) {
  __shared__ __bf16 aT[2][64 * 32];
  __shared__ __bf16 bT[2][128 * 32];
  const int tid  = threadIdx.x;
  const int lane = tid & 31;
  const int wid  = tid >> 5;
  const int wm   = wid >> 2, wn = wid & 3;
  const int rBase = blockIdx.x * 64;
  const int nBase = blockIdx.y * 128;       // N = 512 -> 4 blocks, no tail
  const int half = lane >> 4, nr = lane & 15;

  const int arow = tid >> 2, aseg = (tid & 3) * 8;
  const int brow = tid >> 1, bseg = (tid & 1) * 16;

  v8f acc[2][2] = {{zero8(), zero8()}, {zero8(), zero8()}};

  auto stage = [&](int k0, int buf) {
    // Both tiles already bf16 in memory -> pure async DMA staging.
    async_ld_b128(&aT[buf][arow * 32 + aseg],
                  ynorm + (size_t)(rBase + arow) * DINNER + k0 + aseg);
    const __bf16* bsrc = Wo + (size_t)(nBase + brow) * DINNER + k0 + bseg;
    async_ld_b128(&bT[buf][brow * 32 + bseg], bsrc);
    async_ld_b128(&bT[buf][brow * 32 + bseg + 8], bsrc + 8);
    if (k0 + 32 < DINNER)
      __builtin_prefetch(bsrc + 32, 0, 1);
  };

  stage(0, 0);
  const int NK = DINNER / 32;               // 32
  for (int kt = 0; kt < NK; ++kt) {
    wait_async0();
    __syncthreads();
    if (kt + 1 < NK) stage((kt + 1) * 32, (kt + 1) & 1);
    const __bf16* aB = aT[kt & 1];
    const __bf16* bB = bT[kt & 1];
    v16bf af0 = load_a_frag(aB, wm * 32,      lane, 32, 0);
    v16bf af1 = load_a_frag(aB, wm * 32 + 16, lane, 32, 0);
    v16bf bf0 = load_b_frag(bB, wn * 32,      lane, 32, 0);
    v16bf bf1 = load_b_frag(bB, wn * 32 + 16, lane, 32, 0);
    acc[0][0] = wmma_bf16(af0, bf0, acc[0][0]);
    acc[0][1] = wmma_bf16(af0, bf1, acc[0][1]);
    acc[1][0] = wmma_bf16(af1, bf0, acc[1][0]);
    acc[1][1] = wmma_bf16(af1, bf1, acc[1][1]);
  }
#pragma unroll
  for (int mi = 0; mi < 2; ++mi)
#pragma unroll
    for (int ni = 0; ni < 2; ++ni)
#pragma unroll
      for (int i = 0; i < 8; ++i) {
        int r  = rBase + wm * 32 + mi * 16 + half * 8 + i;
        int gn = nBase + wn * 32 + ni * 16 + nr;
        int t = r & 7, w = (r >> 3) & 15, hh = r >> 7;
        out[((size_t)((t * 16 + hh) * 16 + w) << 9) + gn] = acc[mi][ni][i];
      }
}

// ================= Launch =================
extern "C" void kernel_launch(void* const* d_in, const int* in_sizes, int n_in,
                              void* d_out, int out_size, void* d_ws, size_t ws_size,
                              hipStream_t stream) {
  const float* x          = (const float*)d_in[0];
  const float* in_proj_w  = (const float*)d_in[1];
  const float* conv_w     = (const float*)d_in[2];
  const float* conv_b     = (const float*)d_in[3];
  const float* dt_bias    = (const float*)d_in[4];
  const float* A_log      = (const float*)d_in[5];
  const float* Dvec       = (const float*)d_in[6];
  const float* norm_w     = (const float*)d_in[7];
  const float* out_proj_w = (const float*)d_in[8];
  float* out = (float*)d_out;

  char* ws = (char*)d_ws;
  size_t off = 0;
  auto carve = [&](size_t bytes) -> void* {
    void* p = ws + off;
    off += (bytes + 255) & ~(size_t)255;
    return p;
  };
  float*  zx    = (float*)carve((size_t)NROWS * DINPROJ * sizeof(float));
  float*  xBC   = (float*)carve((size_t)NROWS * CONVDIM * sizeof(float));
  float*  dtb   = (float*)carve((size_t)NROWS * NHEADS * sizeof(float));
  float*  dAb   = (float*)carve((size_t)NROWS * NHEADS * sizeof(float));
  float*  ybuf  = (float*)carve((size_t)NROWS * DINNER * sizeof(float));
  __bf16* ynorm = (__bf16*)carve((size_t)NROWS * DINNER * sizeof(__bf16));
  __bf16* Wi    = (__bf16*)carve((size_t)DINPROJ_PAD * DMODEL * sizeof(__bf16));
  __bf16* Wo    = (__bf16*)carve((size_t)DMODEL * DINNER * sizeof(__bf16));

  // One-shot bf16 weight conversion (in_proj padded to 2432 zero rows).
  k0_cvt<<<(DINPROJ_PAD * DMODEL) / 8 / 256, 256, 0, stream>>>(
      in_proj_w, Wi, DINPROJ * DMODEL, DINPROJ_PAD * DMODEL);
  k0_cvt<<<(DMODEL * DINNER) / 8 / 256, 256, 0, stream>>>(
      out_proj_w, Wo, DMODEL * DINNER, DMODEL * DINNER);

  k1_inproj<<<dim3(NROWS / 64, DINPROJ_PAD / 128), 256, 0, stream>>>(x, Wi, zx);
  k2_conv<<<(BFLAT * CONVDIM) / 256, 256, 0, stream>>>(zx, conv_w, conv_b, xBC);
  k2b_dt<<<(NROWS * NHEADS) / 256, 256, 0, stream>>>(zx, dt_bias, A_log, dtb, dAb);
  k3_ssd<<<(BFLAT * NHEADS) / 4, 128, 0, stream>>>(xBC, dtb, dAb, Dvec, ybuf);
  k4_gatenorm<<<NROWS, 256, 0, stream>>>(zx, ybuf, norm_w, ynorm);
  k5_outproj<<<dim3(NROWS / 64, DMODEL / 128), 256, 0, stream>>>(ynorm, Wo, out);
}
